// Mamba2_10565619548341
// MI455X (gfx1250) — compile-verified, matrix operand order checked by reading, not run on hardware
//
#include <hip/hip_runtime.h>

#define SEQ      2048
#define D_MODEL  2048
#define D_INNER  4096
#define D_XB     512
#define D_STATE  16
#define DT_RANK  128
#define ZXBC_N   (2*D_XB + 2*D_INNER)   // 9216
#define NUM_HEAD (D_INNER / D_STATE)    // 256
#define TCH      128                    // scan time-chunk staged in LDS

typedef __attribute__((ext_vector_type(16))) __bf16 v16bf;
typedef __attribute__((ext_vector_type(8)))  float  v8f;

__device__ __forceinline__ float silu_f(float x) { return x / (1.0f + __expf(-x)); }
__device__ __forceinline__ float softplus_f(float x) {
  return (x > 20.0f) ? x : log1pf(__expf(x));
}

// Load 16 bf16 values for one lane of a 16x32 (or 32x16) WMMA fragment.
// ISA 7.12.2 (16-bit A/B 16x32): lane holds K = koff+0..7 (elems 0..7) and
// K = koff+16..23 (elems 8..15), koff = (lane<16 ? 0 : 8). `p` points at
// row_base + koff already.
__device__ __forceinline__ v16bf load_frag_bf16(const float* __restrict__ p) {
  v16bf r;
#pragma unroll
  for (int i = 0; i < 8; ++i) r[i]     = (__bf16)p[i];
#pragma unroll
  for (int i = 0; i < 8; ++i) r[8 + i] = (__bf16)p[16 + i];
  return r;
}

__device__ __forceinline__ v8f wmma_bf16(v16bf a, v16bf b, v8f c) {
  return __builtin_amdgcn_wmma_f32_16x16x32_bf16(false, a, false, b, (short)0,
                                                 c, false, false);
}

// C[M,N] = A[M,K] @ W[N,K]^T   (all fp32 row-major; bf16 WMMA, f32 accumulate)
// Block = 256 threads = 8 waves arranged 4(M) x 2(N): block tile 128 x 128.
// Each wave computes a 32x64 strip: 2 A-fragments x 4 B-fragments = 8 WMMAs
// per 32-wide k-step (3 b128 loads + 6 cvt per WMMA).
// Requires: M % 128 == 0, N % 128 == 0, K % 32 == 0 (true for all calls here).
__global__ __launch_bounds__(256) void gemm_bf16_wmma(
    float* __restrict__ C, const float* __restrict__ A, const float* __restrict__ W,
    int M, int N, int K) {
  const int lane = threadIdx.x & 31;
  const int wave = threadIdx.x >> 5;
  const int m0   = blockIdx.x * 128 + (wave & 3) * 32;
  const int n0   = blockIdx.y * 128 + (wave >> 2) * 64;
  const int ml   = lane & 15;               // M (for A) / N (for B) within tile
  const int koff = (lane < 16) ? 0 : 8;

  const float* arow0 = A + (size_t)(m0 + ml) * K + koff;
  const float* arow1 = A + (size_t)(m0 + 16 + ml) * K + koff;
  const float* wrow0 = W + (size_t)(n0 + 0 * 16 + ml) * K + koff;
  const float* wrow1 = W + (size_t)(n0 + 1 * 16 + ml) * K + koff;
  const float* wrow2 = W + (size_t)(n0 + 2 * 16 + ml) * K + koff;
  const float* wrow3 = W + (size_t)(n0 + 3 * 16 + ml) * K + koff;

  v8f z = {0.f,0.f,0.f,0.f,0.f,0.f,0.f,0.f};
  v8f acc00 = z, acc01 = z, acc02 = z, acc03 = z;
  v8f acc10 = z, acc11 = z, acc12 = z, acc13 = z;

  for (int kk = 0; kk < K; kk += 32) {
    // prefetch the next k-slab of the streamed rows
    __builtin_prefetch(arow0 + kk + 32, 0, 1);
    __builtin_prefetch(wrow0 + kk + 32, 0, 1);
    __builtin_prefetch(wrow2 + kk + 32, 0, 1);

    v16bf a0 = load_frag_bf16(arow0 + kk);
    v16bf a1 = load_frag_bf16(arow1 + kk);

    v16bf b0 = load_frag_bf16(wrow0 + kk);
    acc00 = wmma_bf16(a0, b0, acc00);
    acc10 = wmma_bf16(a1, b0, acc10);
    v16bf b1 = load_frag_bf16(wrow1 + kk);
    acc01 = wmma_bf16(a0, b1, acc01);
    acc11 = wmma_bf16(a1, b1, acc11);
    v16bf b2 = load_frag_bf16(wrow2 + kk);
    acc02 = wmma_bf16(a0, b2, acc02);
    acc12 = wmma_bf16(a1, b2, acc12);
    v16bf b3 = load_frag_bf16(wrow3 + kk);
    acc03 = wmma_bf16(a0, b3, acc03);
    acc13 = wmma_bf16(a1, b3, acc13);
  }

  // C/D layout: VGPR r -> M = r (lanes 0-15) or M = 8+r (lanes 16-31); N = lane%16
  const int rb0 = m0 + ((lane >> 4) << 3);
#pragma unroll
  for (int r = 0; r < 8; ++r) {
    size_t row = (size_t)(rb0 + r) * N + n0 + ml;
    C[row +  0] = acc00[r];
    C[row + 16] = acc01[r];
    C[row + 32] = acc02[r];
    C[row + 48] = acc03[r];
    size_t row2 = (size_t)(rb0 + 16 + r) * N + n0 + ml;
    C[row2 +  0] = acc10[r];
    C[row2 + 16] = acc11[r];
    C[row2 + 32] = acc12[r];
    C[row2 + 48] = acc13[r];
  }
}

// u[t,d] = silu(conv_b[d] + sum_j xh[t-3+j, d] * conv_w[d,j]),
// xh[t,d] = zxbc[t, D_INNER + (d/128)*16 + d%16]   (x slice with head-repeat)
__global__ __launch_bounds__(256) void conv_silu_kernel(
    const float* __restrict__ zxbc, const float* __restrict__ conv_w,
    const float* __restrict__ conv_b, float* __restrict__ ubuf) {
  int idx = blockIdx.x * 256 + threadIdx.x;      // t * D_INNER + d
  int t = idx >> 12;
  int d = idx & (D_INNER - 1);
  int xcol = D_INNER + ((d >> 7) << 4) + (d & 15);
  float acc = conv_b[d];
#pragma unroll
  for (int j = 0; j < 4; ++j) {
    int ts = t - 3 + j;
    if (ts >= 0) acc += zxbc[(size_t)ts * ZXBC_N + xcol] * conv_w[d * 4 + j];
  }
  ubuf[idx] = silu_f(acc);
}

// Selective scan: one block per head (256 blocks x 256 threads).
// Thread (p = tid>>4, n = tid&15) owns state element h[p][n].
// Chunks of TCH timesteps staged through LDS; softplus/gating fused.
__global__ __launch_bounds__(256) void scan_kernel(
    const float* __restrict__ zxbc, const float* __restrict__ dtbuf,
    const float* __restrict__ ubuf, const float* __restrict__ dt_bias,
    const float* __restrict__ A_log, const float* __restrict__ Dvec,
    float* __restrict__ ybuf) {
  __shared__ float s_dt[TCH][16];
  __shared__ float s_u [TCH][16];
  __shared__ float s_B [TCH][16];
  __shared__ float s_C [TCH][16];
  __shared__ float s_z [TCH][16];

  const int head  = blockIdx.x;           // 0..255
  const int bhead = head >> 3;            // source head for x/B repeat
  const int tid   = threadIdx.x;
  const int p     = tid >> 4;
  const int n     = tid & 15;

  const float Aneg = -__expf(A_log[(head * 16 + p) * 16 + n]);
  const float Dv   = Dvec[head * 16 + p];
  float h = 0.0f;

  for (int t0 = 0; t0 < SEQ; t0 += TCH) {
    // cooperative staging: each array TCH*16 = 2048 elements, 8 per thread
    for (int i = tid; i < TCH * 16; i += 256) {
      int tt = i >> 4, c = i & 15;
      int tg = t0 + tt;
      size_t zrow = (size_t)tg * ZXBC_N;
      s_dt[tt][c] = softplus_f(dtbuf[(size_t)tg * D_INNER + head * 16 + c] +
                               dt_bias[head * 16 + c]);
      s_u [tt][c] = ubuf[(size_t)tg * D_INNER + head * 16 + c];
      s_B [tt][c] = zxbc[zrow + D_INNER + D_XB + bhead * 16 + c];
      s_C [tt][c] = zxbc[zrow + D_INNER + 2 * D_XB + head * 16 + c];
      s_z [tt][c] = zxbc[zrow + head * 16 + c];
    }
    __syncthreads();

    for (int tt = 0; tt < TCH; ++tt) {
      float dtv = s_dt[tt][p];
      float dA  = __expf(dtv * Aneg);
      h = h * dA + dtv * s_u[tt][p] * s_B[tt][n];
      float yv = h * s_C[tt][n];
      // reduce over n within 16-lane groups
      yv += __shfl_xor(yv, 8, 32);
      yv += __shfl_xor(yv, 4, 32);
      yv += __shfl_xor(yv, 2, 32);
      yv += __shfl_xor(yv, 1, 32);
      if (n == 0) {
        float uu = s_u[tt][p];
        float zz = s_z[tt][p];
        ybuf[(size_t)(t0 + tt) * D_INNER + head * 16 + p] =
            (yv + uu * Dv) * silu_f(zz);
      }
    }
    __syncthreads();
  }
}

extern "C" void kernel_launch(void* const* d_in, const int* in_sizes, int n_in,
                              void* d_out, int out_size, void* d_ws, size_t ws_size,
                              hipStream_t stream) {
  (void)in_sizes; (void)n_in; (void)out_size; (void)ws_size;
  const float* hs      = (const float*)d_in[0];  // (1, 2048, 2048)
  const float* in_w    = (const float*)d_in[1];  // (9216, 2048)
  const float* dtin_w  = (const float*)d_in[2];  // (128, 2048)
  const float* dtp_w   = (const float*)d_in[3];  // (4096, 128)
  const float* dtp_b   = (const float*)d_in[4];  // (4096,)
  const float* conv_w  = (const float*)d_in[5];  // (4096, 1, 4)
  const float* conv_b  = (const float*)d_in[6];  // (4096,)
  const float* A_log   = (const float*)d_in[7];  // (4096, 16)
  const float* Dvec    = (const float*)d_in[8];  // (4096,)
  const float* out_w   = (const float*)d_in[9];  // (2048, 4096)
  float* out = (float*)d_out;                    // (1, 2048, 2048) fp32

  float* ws = (float*)d_ws;
  size_t off = 0;
  float* zxbc = ws + off; off += (size_t)SEQ * ZXBC_N;    // 2048 x 9216
  float* dtl  = ws + off; off += (size_t)SEQ * DT_RANK;   // 2048 x 128
  float* dtb  = ws + off; off += (size_t)SEQ * D_INNER;   // 2048 x 4096
  float* ub   = ws + off; off += (size_t)SEQ * D_INNER;   // 2048 x 4096
  float* yb   = ws + off; off += (size_t)SEQ * D_INNER;   // 2048 x 4096

  dim3 blk(256);

  // zxbc = hs @ in_proj_w.T
  gemm_bf16_wmma<<<dim3(SEQ / 128, ZXBC_N / 128), blk, 0, stream>>>(
      zxbc, hs, in_w, SEQ, ZXBC_N, D_MODEL);
  // dtl = hs @ dt_in_proj_w.T
  gemm_bf16_wmma<<<dim3(SEQ / 128, DT_RANK / 128), blk, 0, stream>>>(
      dtl, hs, dtin_w, SEQ, DT_RANK, D_MODEL);
  // dtb = dtl @ dt_proj_w.T
  gemm_bf16_wmma<<<dim3(SEQ / 128, D_INNER / 128), blk, 0, stream>>>(
      dtb, dtl, dtp_w, SEQ, D_INNER, DT_RANK);
  // depthwise causal conv + silu -> u
  conv_silu_kernel<<<(SEQ * D_INNER) / 256, blk, 0, stream>>>(
      zxbc, conv_w, conv_b, ub);
  // selective scan + gating -> yb
  scan_kernel<<<NUM_HEAD, blk, 0, stream>>>(
      zxbc, dtb, ub, dtp_b, A_log, Dvec, yb);
  // out = yb @ out_proj_w.T
  gemm_bf16_wmma<<<dim3(SEQ / 128, D_MODEL / 128), blk, 0, stream>>>(
      out, yb, out_w, SEQ, D_MODEL, D_INNER);
}
